// QLLKTransformer_3736621547701
// MI455X (gfx1250) — matrix operand order repaired and checked by reading, not compiled, not run on hardware
//
#include <hip/hip_runtime.h>
#include <hip/hip_bf16.h>
#include <math.h>

// ---------------------------------------------------------------------------
// Types for CDNA5 WMMA (wave32)
// ---------------------------------------------------------------------------
typedef __attribute__((ext_vector_type(16))) __bf16 v16bf;
typedef __attribute__((ext_vector_type(8)))  __bf16 v8bf;
typedef __attribute__((ext_vector_type(8)))  float  v8f;

// Model constants
#define DIMD     512
#define NLAYERS  4
#define PATCH    8
#define VOCAB    256
#define BATCH    4
#define SEQ      32768
#define NPATCH   (SEQ / PATCH)          // 4096
#define MROWS    (BATCH * NPATCH)       // 16384
#define HIDDEN   (2 * DIMD)             // 1024
#define DEC_N    (VOCAB * PATCH)        // 2048
#define EPSLN    1e-5f

// Epilogue modes
#define EPI_NONE      0   // outF = acc + bias
#define EPI_SIGMOID   1   // outF = sigmoid(acc + bias)
#define EPI_TANH_RES  2   // o = resid + tanh(acc + bias); outF = o; outB = bf16(o)
#define EPI_GELU_B    3   // o = gelu_exact(acc + bias);   outB = bf16(o)

// ---------------------------------------------------------------------------
// Fragment loaders (layouts per CDNA5 ISA 7.12.2, bf16 16x16x32)
//   A 16x32 : lane L holds row M=(L&15); k-base = (L>>4)*8; elems 0..7 -> k..k+7,
//             elems 8..15 -> k+16..k+23  (two contiguous 16B loads)
//   B 32x16 : lane L holds col N=(L&15); same k order as A (strided loads)
// ---------------------------------------------------------------------------
__device__ __forceinline__ v16bf load_a_frag(const __bf16* A, int lda, int m0, int k0) {
    unsigned lane = threadIdx.x & 31u;
    int row = lane & 15;
    int kb  = (lane >> 4) * 8;
    const __bf16* p = A + (size_t)(m0 + row) * lda + (k0 + kb);
    v8bf lo = *(const v8bf*)p;
    v8bf hi = *(const v8bf*)(p + 16);
    v16bf r;
#pragma unroll
    for (int i = 0; i < 8; ++i) { r[i] = lo[i]; r[i + 8] = hi[i]; }
    return r;
}

__device__ __forceinline__ v16bf load_b_frag(const __bf16* Bw, int ldb, int k0, int n0) {
    unsigned lane = threadIdx.x & 31u;
    int kb = (int)(lane >> 4) * 8;
    const __bf16* p = Bw + (size_t)(k0 + kb) * ldb + n0 + (lane & 15);
    v16bf r;
#pragma unroll
    for (int i = 0; i < 8; ++i) { r[i] = p[(size_t)i * ldb]; r[i + 8] = p[(size_t)(i + 16) * ldb]; }
    return r;
}

// ---------------------------------------------------------------------------
// GEMM: C[M,N] = act(A[M,K] * B[K,N] + bias) ; A,B bf16 row-major, acc f32.
// Block = 256 threads = 8 waves; block tile 128(M) x 64(N); wave tile 32x32.
// grid = (N/64, M/128)
// ---------------------------------------------------------------------------
template <int EPI>
__global__ __launch_bounds__(256)
void gemm_bf16_wmma(const __bf16* __restrict__ A, const __bf16* __restrict__ Bw,
                    const float* __restrict__ bias, const float* __restrict__ resid,
                    float* __restrict__ outF, __bf16* __restrict__ outB,
                    int M, int N, int K) {
    unsigned wave = threadIdx.x >> 5;
    unsigned lane = threadIdx.x & 31u;
    int m0 = blockIdx.y * 128 + (int)(wave >> 1) * 32;
    int n0 = blockIdx.x * 64  + (int)(wave & 1)  * 32;

    v8f acc00 = {}, acc01 = {}, acc10 = {}, acc11 = {};

    for (int k0 = 0; k0 < K; k0 += 32) {
        v16bf a0 = load_a_frag(A, K, m0,      k0);
        v16bf a1 = load_a_frag(A, K, m0 + 16, k0);
        v16bf b0 = load_b_frag(Bw, N, k0, n0);
        v16bf b1 = load_b_frag(Bw, N, k0, n0 + 16);
        acc00 = __builtin_amdgcn_wmma_f32_16x16x32_bf16(false, a0, false, b0, (short)0, acc00, false, false);
        acc01 = __builtin_amdgcn_wmma_f32_16x16x32_bf16(false, a0, false, b1, (short)0, acc01, false, false);
        acc10 = __builtin_amdgcn_wmma_f32_16x16x32_bf16(false, a1, false, b0, (short)0, acc10, false, false);
        acc11 = __builtin_amdgcn_wmma_f32_16x16x32_bf16(false, a1, false, b1, (short)0, acc11, false, false);
    }

    // C/D layout: lane L -> col N=(L&15); VGPR r -> row M = r + 8*(L>>4)
    int colL = lane & 15;
    int rowB = (int)(lane >> 4) * 8;
    v8f accs[2][2] = {{acc00, acc01}, {acc10, acc11}};

#pragma unroll
    for (int mi = 0; mi < 2; ++mi) {
#pragma unroll
        for (int ni = 0; ni < 2; ++ni) {
            int col = n0 + ni * 16 + colL;
            float bs = bias ? bias[col] : 0.0f;
#pragma unroll
            for (int r = 0; r < 8; ++r) {
                int row = m0 + mi * 16 + rowB + r;
                size_t idx = (size_t)row * N + col;
                float v = accs[mi][ni][r] + bs;
                if (EPI == EPI_NONE) {
                    outF[idx] = v;
                } else if (EPI == EPI_SIGMOID) {
                    outF[idx] = 1.0f / (1.0f + __expf(-v));
                } else if (EPI == EPI_TANH_RES) {
                    float o = resid[idx] + tanhf(v);
                    outF[idx] = o;
                    outB[idx] = (__bf16)o;
                } else { // EPI_GELU_B
                    float o = 0.5f * v * (1.0f + erff(v * 0.70710678118654752f));
                    outB[idx] = (__bf16)o;
                }
            }
        }
    }
}

// ---------------------------------------------------------------------------
// f32 -> bf16 elementwise (weights / activations)
// ---------------------------------------------------------------------------
__global__ void f32_to_bf16_k(const float* __restrict__ in, __bf16* __restrict__ out, int n) {
    int i = blockIdx.x * blockDim.x + threadIdx.x;
    if (i < n) out[i] = (__bf16)in[i];
}

// ---------------------------------------------------------------------------
// Byte embedding gather + patch mean pool. One block per patch row.
// ---------------------------------------------------------------------------
__global__ __launch_bounds__(128)
void embed_pool_k(const int* __restrict__ bytes_in, const float* __restrict__ emb,
                  float* __restrict__ P, __bf16* __restrict__ XB) {
    int row = blockIdx.x;            // b*NPATCH + p
    __shared__ int ids[PATCH];
    if (threadIdx.x < PATCH) ids[threadIdx.x] = bytes_in[(size_t)row * PATCH + threadIdx.x];
    __syncthreads();
#pragma unroll
    for (int c = 0; c < 4; ++c) {
        int dcol = threadIdx.x + c * 128;
        float s = 0.0f;
#pragma unroll
        for (int j = 0; j < PATCH; ++j) s += emb[(size_t)ids[j] * DIMD + dcol];
        s *= (1.0f / PATCH);
        size_t idx = (size_t)row * DIMD + dcol;
        P[idx]  = s;
        XB[idx] = (__bf16)s;
    }
}

// ---------------------------------------------------------------------------
// kv_state = cumsum(k*v) along patch axis; y = x + q*kv*g.
// One thread per (batch, dim) column: 2048 independent scans, coalesced.
// ---------------------------------------------------------------------------
__global__ __launch_bounds__(256)
void kv_scan_k(const float* __restrict__ xf, const float* __restrict__ q,
               const float* __restrict__ k, const float* __restrict__ v,
               const float* __restrict__ g, float* __restrict__ y) {
    int t = blockIdx.x * blockDim.x + threadIdx.x;   // 0 .. B*DIMD-1
    int b = t >> 9;            // /512
    int d = t & (DIMD - 1);
    size_t base = ((size_t)b * NPATCH) * DIMD + d;
    float run = 0.0f;
    for (int p = 0; p < NPATCH; ++p) {
        size_t i = base + (size_t)p * DIMD;
        run += k[i] * v[i];
        y[i] = xf[i] + q[i] * run * g[i];
    }
}

// ---------------------------------------------------------------------------
// LayerNorm over last dim (512). One block (256 thr) per row, 2 elems/thread.
// Writes f32 (residual path) and bf16 (next GEMM A).
// ---------------------------------------------------------------------------
__global__ __launch_bounds__(256)
void layernorm_k(const float* __restrict__ in, const float* __restrict__ sc,
                 const float* __restrict__ bi, float* __restrict__ outF,
                 __bf16* __restrict__ outB) {
    __shared__ float red[256];
    int row = blockIdx.x;
    int t = threadIdx.x;
    const float* x = in + (size_t)row * DIMD;
    float a = x[t], b = x[t + 256];

    red[t] = a + b;
    __syncthreads();
    for (int s = 128; s > 0; s >>= 1) { if (t < s) red[t] += red[t + s]; __syncthreads(); }
    float mu = red[0] * (1.0f / DIMD);
    __syncthreads();

    float da = a - mu, db = b - mu;
    red[t] = da * da + db * db;
    __syncthreads();
    for (int s = 128; s > 0; s >>= 1) { if (t < s) red[t] += red[t + s]; __syncthreads(); }
    float inv = rsqrtf(red[0] * (1.0f / DIMD) + EPSLN);

    float o0 = da * inv * sc[t]       + bi[t];
    float o1 = db * inv * sc[t + 256] + bi[t + 256];
    size_t idx = (size_t)row * DIMD + t;
    outF[idx]       = o0;  outB[idx]       = (__bf16)o0;
    outF[idx + 256] = o1;  outB[idx + 256] = (__bf16)o1;
}

// ---------------------------------------------------------------------------
// Host launcher
// ---------------------------------------------------------------------------
extern "C" void kernel_launch(void* const* d_in, const int* in_sizes, int n_in,
                              void* d_out, int out_size, void* d_ws, size_t ws_size,
                              hipStream_t stream) {
    (void)in_sizes; (void)n_in; (void)out_size; (void)ws_size;

    const int*   bytes_in  = (const int*)  d_in[0];
    const float* emb       = (const float*)d_in[1];
    const float* hash_proj = (const float*)d_in[2];
    const float* Wq = (const float*)d_in[3];  const float* bq = (const float*)d_in[4];
    const float* Wk = (const float*)d_in[5];  const float* bk = (const float*)d_in[6];
    const float* Wv = (const float*)d_in[7];  const float* bv = (const float*)d_in[8];
    const float* Wg = (const float*)d_in[9];  const float* bg = (const float*)d_in[10];
    const float* ln1_s = (const float*)d_in[11]; const float* ln1_b = (const float*)d_in[12];
    const float* W1 = (const float*)d_in[13]; const float* b1 = (const float*)d_in[14];
    const float* W2 = (const float*)d_in[15]; const float* b2 = (const float*)d_in[16];
    const float* ln2_s = (const float*)d_in[17]; const float* ln2_b = (const float*)d_in[18];
    const float* Wd = (const float*)d_in[19]; const float* bd = (const float*)d_in[20];
    float* out = (float*)d_out;

    // ---- workspace carve-up (256B aligned) ----
    char* w = (char*)d_ws;
    auto alloc = [&](size_t bytes) -> void* {
        void* p = (void*)w;
        w += (bytes + 255) & ~(size_t)255;
        return p;
    };
    const size_t D2  = (size_t)DIMD * DIMD;          // 262144
    const size_t ACT = (size_t)MROWS * DIMD;         // 8.39M
    const size_t HID = (size_t)MROWS * HIDDEN;       // 16.8M

    __bf16* WB_HASH = (__bf16*)alloc(D2 * 2);
    __bf16* WB_WQ   = (__bf16*)alloc(NLAYERS * D2 * 2);
    __bf16* WB_WK   = (__bf16*)alloc(NLAYERS * D2 * 2);
    __bf16* WB_WV   = (__bf16*)alloc(NLAYERS * D2 * 2);
    __bf16* WB_WG   = (__bf16*)alloc(NLAYERS * D2 * 2);
    __bf16* WB_W1   = (__bf16*)alloc((size_t)NLAYERS * DIMD * HIDDEN * 2);
    __bf16* WB_W2   = (__bf16*)alloc((size_t)NLAYERS * HIDDEN * DIMD * 2);
    __bf16* WB_WD   = (__bf16*)alloc((size_t)DIMD * DEC_N * 2);

    float*  P   = (float*) alloc(ACT * 4);   // patch_lat (hash residual)
    float*  XF  = (float*) alloc(ACT * 4);   // current x, f32
    float*  Y   = (float*) alloc(ACT * 4);   // pre-LN scratch
    float*  Q   = (float*) alloc(ACT * 4);
    float*  Kb  = (float*) alloc(ACT * 4);
    float*  V   = (float*) alloc(ACT * 4);
    float*  G   = (float*) alloc(ACT * 4);
    __bf16* XB0 = (__bf16*)alloc(ACT * 2);   // ping
    __bf16* XB1 = (__bf16*)alloc(ACT * 2);   // pong
    __bf16* HB  = (__bf16*)alloc(HID * 2);   // FFN hidden (bf16)

    auto conv = [&](const float* src, __bf16* dst, size_t n) {
        f32_to_bf16_k<<<dim3((unsigned)((n + 255) / 256)), 256, 0, stream>>>(src, dst, (int)n);
    };

    // ---- convert all weights to bf16 (L2-resident working set) ----
    conv(hash_proj, WB_HASH, D2);
    conv(Wq, WB_WQ, NLAYERS * D2);
    conv(Wk, WB_WK, NLAYERS * D2);
    conv(Wv, WB_WV, NLAYERS * D2);
    conv(Wg, WB_WG, NLAYERS * D2);
    conv(W1, WB_W1, (size_t)NLAYERS * DIMD * HIDDEN);
    conv(W2, WB_W2, (size_t)NLAYERS * HIDDEN * DIMD);
    conv(Wd, WB_WD, (size_t)DIMD * DEC_N);

    // ---- embedding gather + patch mean ----
    embed_pool_k<<<dim3(MROWS), 128, 0, stream>>>(bytes_in, emb, P, XB0);

    dim3 blk(256);
    dim3 grd_d (DIMD  / 64, MROWS / 128);   // N=512
    dim3 grd_h (HIDDEN/ 64, MROWS / 128);   // N=1024
    dim3 grd_dc(DEC_N / 64, MROWS / 128);   // N=2048

    // ---- x = patch_lat + tanh(patch_lat @ hash_proj) ----
    gemm_bf16_wmma<EPI_TANH_RES><<<grd_d, blk, 0, stream>>>(
        XB0, WB_HASH, nullptr, P, XF, XB1, MROWS, DIMD, DIMD);

    __bf16* xb = XB1;     // bf16 copy of current x
    for (int i = 0; i < NLAYERS; ++i) {
        const __bf16* wq = WB_WQ + (size_t)i * D2;
        const __bf16* wk = WB_WK + (size_t)i * D2;
        const __bf16* wv = WB_WV + (size_t)i * D2;
        const __bf16* wg = WB_WG + (size_t)i * D2;

        gemm_bf16_wmma<EPI_NONE><<<grd_d, blk, 0, stream>>>(
            xb, wq, bq + (size_t)i * DIMD, nullptr, Q,  nullptr, MROWS, DIMD, DIMD);
        gemm_bf16_wmma<EPI_NONE><<<grd_d, blk, 0, stream>>>(
            xb, wk, bk + (size_t)i * DIMD, nullptr, Kb, nullptr, MROWS, DIMD, DIMD);
        gemm_bf16_wmma<EPI_NONE><<<grd_d, blk, 0, stream>>>(
            xb, wv, bv + (size_t)i * DIMD, nullptr, V,  nullptr, MROWS, DIMD, DIMD);
        gemm_bf16_wmma<EPI_SIGMOID><<<grd_d, blk, 0, stream>>>(
            xb, wg, bg + (size_t)i * DIMD, nullptr, G,  nullptr, MROWS, DIMD, DIMD);

        // y = x + q * cumsum(k*v) * g   (2048 independent column scans)
        kv_scan_k<<<dim3((BATCH * DIMD) / 256), 256, 0, stream>>>(XF, Q, Kb, V, G, Y);

        // LN1 -> x (f32 + bf16 ping)
        layernorm_k<<<dim3(MROWS), 256, 0, stream>>>(
            Y, ln1_s + (size_t)i * DIMD, ln1_b + (size_t)i * DIMD, XF, XB0);

        // FFN: h = gelu(x @ W1 + b1)  -> bf16 hidden
        gemm_bf16_wmma<EPI_GELU_B><<<grd_h, blk, 0, stream>>>(
            XB0, WB_W1 + (size_t)i * DIMD * HIDDEN, b1 + (size_t)i * HIDDEN,
            nullptr, nullptr, HB, MROWS, HIDDEN, DIMD);

        // x = h @ W2 + b2
        gemm_bf16_wmma<EPI_NONE><<<grd_d, blk, 0, stream>>>(
            HB, WB_W2 + (size_t)i * HIDDEN * DIMD, b2 + (size_t)i * DIMD,
            nullptr, Y, nullptr, MROWS, DIMD, HIDDEN);

        // LN2 -> x (f32 + bf16 pong)
        layernorm_k<<<dim3(MROWS), 256, 0, stream>>>(
            Y, ln2_s + (size_t)i * DIMD, ln2_b + (size_t)i * DIMD, XF, XB1);
        xb = XB1;
    }

    // ---- decoder: logits = x @ Wd + bd  ([16384, 2048] == [B, 32768, 256] flat) ----
    gemm_bf16_wmma<EPI_NONE><<<grd_dc, blk, 0, stream>>>(
        xb, WB_WD, bd, nullptr, out, nullptr, MROWS, DEC_N, DIMD);
}
